// SwinTransformerBlock_12266426597634
// MI455X (gfx1250) — matrix operand order repaired and checked by reading, not compiled, not run on hardware
//
#include <hip/hip_runtime.h>
#include <hip/hip_bf16.h>
#include <stdint.h>

// ---------------------------------------------------------------------------
// Swin Transformer block on gfx1250 (CDNA5, wave32, WMMA f16 -> f32)
// B=32, H=W=64, C=192, WS=8, SS=4, NH=6, N=64, nW=64, Dh=32
// ---------------------------------------------------------------------------

typedef __attribute__((ext_vector_type(16))) _Float16 v16h;
typedef __attribute__((ext_vector_type(8)))  float    v8f;

union Frag16 { v16h h; uint32_t u[8]; };

#define M_TOTAL   131072      // B*H*W tokens
#define C_DIM     192
#define SCALE_ATT 0.17677669529663688f   // 32^-0.5

__device__ __forceinline__ float gelu_exact(float x) {
  return 0.5f * x * (1.0f + erff(x * 0.70710678118654752f));
}
// region index along one axis for the shifted-window mask (H=64, WS=8, SS=4)
__device__ __forceinline__ int swin_region(int p) {
  return (p < 56) ? 0 : ((p < 60) ? 1 : 2);
}

// ---------------------------------------------------------------------------
// Weight transpose + f32->f16 convert:  WT[n*K + k] = W[k*N + n]
// ---------------------------------------------------------------------------
__global__ void wt_convert_kernel(const float* __restrict__ W,
                                  _Float16* __restrict__ WT, int K, int N) {
  int idx = blockIdx.x * 256 + threadIdx.x;
  if (idx >= K * N) return;
  int n = idx / K, k = idx % K;
  WT[idx] = (_Float16)W[k * N + n];
}

// ---------------------------------------------------------------------------
// LayerNorm (wave per token). shifted=1: also apply cyclic shift (-4,-4) and
// window partition so output row r is in (B, win, token) order.
// ---------------------------------------------------------------------------
__global__ __launch_bounds__(256) void ln_kernel(
    const float* __restrict__ x, const float* __restrict__ g,
    const float* __restrict__ beta, _Float16* __restrict__ out, int shifted) {
  int wv = threadIdx.x >> 5, lane = threadIdx.x & 31;
  int r = blockIdx.x * 8 + wv;                 // destination row
  size_t src;
  if (shifted) {
    int b = r >> 12, rem = r & 4095;
    int w = rem >> 6, t = rem & 63;
    int hr = ((w >> 3) << 3) + (t >> 3);       // rolled-image row
    int wr = ((w & 7) << 3) + (t & 7);         // rolled-image col
    int hs = (hr + 4) & 63, wsrc = (wr + 4) & 63;  // un-roll: roll(x,-4)
    src = ((size_t)(b << 12) + (hs << 6) + wsrc) * C_DIM;
  } else {
    src = (size_t)r * C_DIM;
  }
  float vals[6];
  float sum = 0.f;
#pragma unroll
  for (int j = 0; j < 6; ++j) { vals[j] = x[src + lane + 32 * j]; sum += vals[j]; }
#pragma unroll
  for (int m = 1; m < 32; m <<= 1) sum += __shfl_xor(sum, m, 32);
  float mean = sum * (1.f / 192.f);
  float vs = 0.f;
#pragma unroll
  for (int j = 0; j < 6; ++j) { float d = vals[j] - mean; vs += d * d; }
#pragma unroll
  for (int m = 1; m < 32; m <<= 1) vs += __shfl_xor(vs, m, 32);
  float rs = rsqrtf(vs * (1.f / 192.f) + 1e-5f);
#pragma unroll
  for (int j = 0; j < 6; ++j) {
    int c = lane + 32 * j;
    out[(size_t)r * C_DIM + c] = (_Float16)((vals[j] - mean) * rs * g[c] + beta[c]);
  }
}

// ---------------------------------------------------------------------------
// Tiled WMMA GEMM: Y = act(X[f16, MxK] * WT[f16, NxK]^T + bias) (+ residual)
// Block tile 128 x (NFRAG*16), 8 waves, each wave 16(M) x NFRAG*16(N) via
// NFRAG WMMA per K-step. NFRAG must be a multiple of 4.
// EPI: 0 = f16 store, 1 = GELU + f16 store,
//      2 = proj: window-reverse+unshift scatter, +x residual, f32 store
//      3 = fc2: += residual (in-place d_out), f32 store
// ---------------------------------------------------------------------------
template <int EPI, int NFRAG>
__global__ __launch_bounds__(256) void gemm_wmma_kernel(
    const _Float16* __restrict__ X, const _Float16* __restrict__ WT,
    const float* __restrict__ bias, void* __restrict__ Yout,
    const float* __restrict__ resid, int M, int N, int K) {
  constexpr int BN = NFRAG * 16;
  __shared__ _Float16 As[128 * 32];
  __shared__ _Float16 Bs[BN * 32];
  const int tid = threadIdx.x;
  const int wv = tid >> 5, lane = tid & 31;
  const int lhalf = lane >> 4, l15 = lane & 15;
  const int m0 = blockIdx.x * 128;
  const int n0 = blockIdx.y * BN;

  v8f acc[NFRAG];
#pragma unroll
  for (int gI = 0; gI < NFRAG; ++gI)
#pragma unroll
    for (int v = 0; v < 8; ++v) acc[gI][v] = 0.f;

  for (int k0 = 0; k0 < K; k0 += 32) {
    __syncthreads();
    {
      const int idx0 = tid * 8;
#pragma unroll
      for (int it = 0; it < 2; ++it) {   // A tile: 128x32 halves
        int idx = idx0 + it * 2048;
        int row = idx >> 5, col = idx & 31;
        *(uint4*)(&As[idx]) =
            *(const uint4*)(&X[(size_t)(m0 + row) * K + k0 + col]);
      }
#pragma unroll
      for (int it = 0; it < NFRAG / 4; ++it) {  // B tile: BNx32 halves
        int idx = idx0 + it * 2048;
        int row = idx >> 5, col = idx & 31;
        *(uint4*)(&Bs[idx]) =
            *(const uint4*)(&WT[(size_t)(n0 + row) * K + k0 + col]);
      }
      // prefetch next A tile into cache while this K-step computes
      if (k0 + 32 < K) {
        int row = idx0 >> 5, col = idx0 & 31;
        __builtin_prefetch(&X[(size_t)(m0 + row) * K + k0 + 32 + col], 0, 3);
      }
    }
    __syncthreads();

    Frag16 a;
    {
      int row = wv * 16 + l15;
#pragma unroll
      for (int v = 0; v < 8; ++v) {
        int kof = ((v & 4) << 2) + (lhalf << 3) + ((v & 3) << 1);
        a.u[v] = *(const uint32_t*)(&As[row * 32 + kof]);
      }
    }
#pragma unroll
    for (int gI = 0; gI < NFRAG; ++gI) {
      Frag16 b;
      int n = gI * 16 + l15;
#pragma unroll
      for (int v = 0; v < 8; ++v) {
        int kof = ((v & 4) << 2) + (lhalf << 3) + ((v & 3) << 1);
        b.u[v] = *(const uint32_t*)(&Bs[n * 32 + kof]);
      }
      acc[gI] = __builtin_amdgcn_wmma_f32_16x16x32_f16(
          false, a.h, false, b.h, (short)0, acc[gI], false, false);
    }
  }

  // epilogue
#pragma unroll
  for (int gI = 0; gI < NFRAG; ++gI) {
#pragma unroll
    for (int v = 0; v < 8; ++v) {
      int row = m0 + wv * 16 + v + (lhalf << 3);
      int col = n0 + gI * 16 + l15;
      float val = acc[gI][v] + bias[col];
      if constexpr (EPI == 1) val = gelu_exact(val);
      if constexpr (EPI == 0 || EPI == 1) {
        ((_Float16*)Yout)[(size_t)row * N + col] = (_Float16)val;
      } else if constexpr (EPI == 2) {
        // window reverse + un-shift: windowed row -> image row, add shortcut x
        int b = row >> 12, rem = row & 4095;
        int w = rem >> 6, t = rem & 63;
        int hr = ((w >> 3) << 3) + (t >> 3);
        int wr = ((w & 7) << 3) + (t & 7);
        int ho = (hr + 4) & 63, wo = (wr + 4) & 63;   // roll(+4,+4)
        size_t prow = (size_t)(b << 12) + (ho << 6) + wo;
        ((float*)Yout)[prow * C_DIM + col] = resid[prow * C_DIM + col] + val;
      } else {  // EPI == 3, in-place residual add (resid == Yout)
        size_t o = (size_t)row * N + col;
        ((float*)Yout)[o] = resid[o] + val;
      }
    }
  }
}

// ---------------------------------------------------------------------------
// Attention: one 128-thread block per (window, head). q@k^T via WMMA,
// analytic rel-pos bias + shift mask, shfl softmax, P@V via WMMA.
// qkv layout: row-major (131072, 576); q|k|v at col sections 0/192/384.
// ---------------------------------------------------------------------------
__global__ __launch_bounds__(128) void attn_wmma_kernel(
    const _Float16* __restrict__ qkv, const float* __restrict__ rpb,
    _Float16* __restrict__ ctx) {
  __shared__ _Float16 P[64 * 64];
  const int blk = blockIdx.x;          // 0 .. 2048*6-1
  const int win = blk / 6, head = blk % 6;
  const int wv = threadIdx.x >> 5, lane = threadIdx.x & 31;
  const int lhalf = lane >> 4, l15 = lane & 15;
  const size_t base = (size_t)win * 64 * 576;
  const int qc = head * 32, kc = 192 + head * 32, vc = 384 + head * 32;

  // ---- scores = q @ k^T  (A = q 16x32, B^T = k rows, already [N][K]) ----
  Frag16 a;
  {
    int row = wv * 16 + l15;
    const _Float16* qrow = qkv + base + (size_t)row * 576 + qc;
#pragma unroll
    for (int v = 0; v < 8; ++v) {
      int kof = ((v & 4) << 2) + (lhalf << 3) + ((v & 3) << 1);
      a.u[v] = *(const uint32_t*)(&qrow[kof]);
    }
  }
  float s[4][8];
#pragma unroll
  for (int gI = 0; gI < 4; ++gI) {
    Frag16 b;
    int n = gI * 16 + l15;
    const _Float16* krow = qkv + base + (size_t)n * 576 + kc;
#pragma unroll
    for (int v = 0; v < 8; ++v) {
      int kof = ((v & 4) << 2) + (lhalf << 3) + ((v & 3) << 1);
      b.u[v] = *(const uint32_t*)(&krow[kof]);
    }
    v8f c;
#pragma unroll
    for (int v = 0; v < 8; ++v) c[v] = 0.f;
    c = __builtin_amdgcn_wmma_f32_16x16x32_f16(false, a.h, false, b.h,
                                               (short)0, c, false, false);
#pragma unroll
    for (int v = 0; v < 8; ++v) s[gI][v] = c[v];
  }

  // ---- scale + rel-pos bias + shift mask ----
  // faithful to reference quirk: mask window index = flat_window // B
  const int mask_w = win >> 5;
  const int n1i = mask_w >> 3, n2i = mask_w & 7;
  int regj[4]; int yj[4], xj[4];
#pragma unroll
  for (int gI = 0; gI < 4; ++gI) {
    int j = gI * 16 + l15;
    yj[gI] = j >> 3; xj[gI] = j & 7;
    regj[gI] = 3 * swin_region(n1i * 8 + yj[gI]) + swin_region(n2i * 8 + xj[gI]);
  }
#pragma unroll
  for (int v = 0; v < 8; ++v) {
    int i = wv * 16 + v + (lhalf << 3);
    int yi = i >> 3, xi = i & 7;
    int regi = 3 * swin_region(n1i * 8 + yi) + swin_region(n2i * 8 + xi);
#pragma unroll
    for (int gI = 0; gI < 4; ++gI) {
      int rpi = (yi - yj[gI] + 7) * 15 + (xi - xj[gI] + 7);
      float bias = rpb[rpi * 6 + head];
      float mk = (regi == regj[gI]) ? 0.f : -100.f;
      s[gI][v] = s[gI][v] * SCALE_ATT + bias + mk;
    }
  }

  // ---- softmax over j (64 values per row: 4 tiles x 16 lanes) ----
#pragma unroll
  for (int v = 0; v < 8; ++v) {
    float m = -1e30f;
#pragma unroll
    for (int gI = 0; gI < 4; ++gI) m = fmaxf(m, s[gI][v]);
#pragma unroll
    for (int msk = 1; msk < 16; msk <<= 1) m = fmaxf(m, __shfl_xor(m, msk, 32));
    float sum = 0.f;
#pragma unroll
    for (int gI = 0; gI < 4; ++gI) { s[gI][v] = __expf(s[gI][v] - m); sum += s[gI][v]; }
#pragma unroll
    for (int msk = 1; msk < 16; msk <<= 1) sum += __shfl_xor(sum, msk, 32);
    float inv = 1.f / sum;
#pragma unroll
    for (int gI = 0; gI < 4; ++gI) s[gI][v] *= inv;
  }

  // ---- stage probs in LDS (f16, row-major 64x64) ----
#pragma unroll
  for (int gI = 0; gI < 4; ++gI)
#pragma unroll
    for (int v = 0; v < 8; ++v) {
      int i = wv * 16 + v + (lhalf << 3);
      int j = gI * 16 + l15;
      P[i * 64 + j] = (_Float16)s[gI][v];
    }
  __syncthreads();

  // ---- ctx = P (64x64) @ V (64x32): 2 K-steps, 2 N-groups ----
  v8f acc2[2];
#pragma unroll
  for (int gg = 0; gg < 2; ++gg)
#pragma unroll
    for (int v = 0; v < 8; ++v) acc2[gg][v] = 0.f;

#pragma unroll
  for (int ks = 0; ks < 2; ++ks) {
    Frag16 pa;
    {
      int row = wv * 16 + l15;
#pragma unroll
      for (int v = 0; v < 8; ++v) {
        int kof = ks * 32 + ((v & 4) << 2) + (lhalf << 3) + ((v & 3) << 1);
        pa.u[v] = *(const uint32_t*)(&P[row * 64 + kof]);
      }
    }
#pragma unroll
    for (int gg = 0; gg < 2; ++gg) {
      Frag16 vb;
      int d = gg * 16 + l15;
#pragma unroll
      for (int v = 0; v < 8; ++v) {
        int k = ks * 32 + ((v & 4) << 2) + (lhalf << 3) + ((v & 3) << 1);
        const _Float16* vp = qkv + base + (size_t)k * 576 + vc + d;
        union { uint32_t u; _Float16 h[2]; } pk;
        pk.h[0] = vp[0];
        pk.h[1] = vp[576];
        vb.u[v] = pk.u;
      }
      acc2[gg] = __builtin_amdgcn_wmma_f32_16x16x32_f16(
          false, pa.h, false, vb.h, (short)0, acc2[gg], false, false);
    }
  }

  // ---- store ctx (131072, 192) f16 ----
#pragma unroll
  for (int gg = 0; gg < 2; ++gg)
#pragma unroll
    for (int v = 0; v < 8; ++v) {
      int i = win * 64 + wv * 16 + v + (lhalf << 3);
      int c = head * 32 + gg * 16 + l15;
      ctx[(size_t)i * C_DIM + c] = (_Float16)acc2[gg][v];
    }
}

// ---------------------------------------------------------------------------
// Host-side orchestration
// ---------------------------------------------------------------------------
extern "C" void kernel_launch(void* const* d_in, const int* in_sizes, int n_in,
                              void* d_out, int out_size, void* d_ws, size_t ws_size,
                              hipStream_t stream) {
  const float* x      = (const float*)d_in[0];
  const float* n1g    = (const float*)d_in[1];
  const float* n1b    = (const float*)d_in[2];
  const float* qkv_w  = (const float*)d_in[3];
  const float* qkv_b  = (const float*)d_in[4];
  const float* proj_w = (const float*)d_in[5];
  const float* proj_b = (const float*)d_in[6];
  const float* rpb    = (const float*)d_in[7];
  const float* n2g    = (const float*)d_in[8];
  const float* n2b    = (const float*)d_in[9];
  const float* fc1_w  = (const float*)d_in[10];
  const float* fc1_b  = (const float*)d_in[11];
  const float* fc2_w  = (const float*)d_in[12];
  const float* fc2_b  = (const float*)d_in[13];
  float* out = (float*)d_out;

  // workspace layout (bytes):
  //   [0, 50.3MB)       hwin f16  (LN1+shift)      -> reused as y2 (LN2 out)
  //   [A, A+201.3MB)    qkv f16 (151MB) + ctx f16 (50.3MB) -> reused as fc1 act
  //   [A+201.3MB, ...)  f16 weights
  char* ws = (char*)d_ws;
  const size_t SZ_HWIN = (size_t)M_TOTAL * C_DIM * 2;          // 50,331,648
  const size_t SZ_QKV  = (size_t)M_TOTAL * 576 * 2;            // 150,994,944
  const size_t SZ_CTX  = SZ_HWIN;
  _Float16* hwin = (_Float16*)ws;
  _Float16* y2   = hwin;                       // reuse after attention done
  _Float16* qkv  = (_Float16*)(ws + SZ_HWIN);
  _Float16* ctx  = (_Float16*)(ws + SZ_HWIN + SZ_QKV);
  _Float16* a1   = qkv;                        // fc1 act reuses qkv+ctx region
  size_t woff = SZ_HWIN + SZ_QKV + SZ_CTX;
  _Float16* qkvT = (_Float16*)(ws + woff); woff += (size_t)576 * 192 * 2;
  _Float16* projT = (_Float16*)(ws + woff); woff += (size_t)192 * 192 * 2;
  _Float16* fc1T = (_Float16*)(ws + woff); woff += (size_t)768 * 192 * 2;
  _Float16* fc2T = (_Float16*)(ws + woff); woff += (size_t)192 * 768 * 2;

  // 1. convert weights to f16 [N][K]
  wt_convert_kernel<<<(192 * 576 + 255) / 256, 256, 0, stream>>>(qkv_w, qkvT, 192, 576);
  wt_convert_kernel<<<(192 * 192 + 255) / 256, 256, 0, stream>>>(proj_w, projT, 192, 192);
  wt_convert_kernel<<<(192 * 768 + 255) / 256, 256, 0, stream>>>(fc1_w, fc1T, 192, 768);
  wt_convert_kernel<<<(768 * 192 + 255) / 256, 256, 0, stream>>>(fc2_w, fc2T, 768, 192);

  // 2. LN1 + shift + window partition -> hwin f16
  ln_kernel<<<M_TOTAL / 8, 256, 0, stream>>>(x, n1g, n1b, hwin, 1);

  // 3. qkv = hwin @ qkv_w + b   (131072 x 576); X fits in L2, BN=64 for
  //    maximum grid parallelism
  gemm_wmma_kernel<0, 4><<<dim3(M_TOTAL / 128, 576 / 64), 256, 0, stream>>>(
      hwin, qkvT, qkv_b, qkv, nullptr, M_TOTAL, 576, 192);

  // 4. windowed attention -> ctx f16 (131072 x 192)
  attn_wmma_kernel<<<2048 * 6, 128, 0, stream>>>(qkv, rpb, ctx);

  // 5. proj + window-reverse/unshift scatter + shortcut -> d_out (x_res, f32)
  //    BN=192: single N-block, ctx read exactly once
  gemm_wmma_kernel<2, 12><<<dim3(M_TOTAL / 128, 1), 256, 0, stream>>>(
      ctx, projT, proj_b, out, x, M_TOTAL, 192, 192);

  // 6. LN2 -> y2 f16
  ln_kernel<<<M_TOTAL / 8, 256, 0, stream>>>(out, n2g, n2b, y2, 0);

  // 7. fc1 + exact GELU -> a1 f16 (131072 x 768); y2 fits in L2, BN=64
  gemm_wmma_kernel<1, 4><<<dim3(M_TOTAL / 128, 768 / 64), 256, 0, stream>>>(
      y2, fc1T, fc1_b, a1, nullptr, M_TOTAL, 768, 192);

  // 8. fc2 + residual (in-place on d_out) -> final output f32
  //    BN=192: a1 (201MB > L2) streamed from HBM exactly once
  gemm_wmma_kernel<3, 12><<<dim3(M_TOTAL / 128, 1), 256, 0, stream>>>(
      a1, fc2T, fc2_b, out, out, M_TOTAL, 192, 768);
}
